// TorchCosineSimilarityLoss_50714973831634
// MI455X (gfx1250) — compile-verified, matrix operand order checked by reading, not run on hardware
//
#include <hip/hip_runtime.h>

typedef __attribute__((ext_vector_type(16))) _Float16 v16h;
typedef __attribute__((ext_vector_type(4)))  _Float16 v4h;
typedef __attribute__((ext_vector_type(8)))  float    v8f;

#define BT   2048
#define HS   2048      // student hidden
#define HT   4096      // teacher hidden
#define VOC  32000
#define ROWS 16        // rows of the batch per workgroup
#define WAVES 8
#define THREADS 256    // 8 wave32s
#define IGNORE_INDEX (-100)

// LDS row strides in f16 elements, padded (+8 halves = 4 dwords) so that the
// 16-lane b128 reads walk banks 0,4,8,...,60 instead of all hitting bank 0.
#define SROW (HS + 8)  // 2056
#define TROW (HT + 8)  // 4104

// ---------------------------------------------------------------------------
// Kernel 1: f32 -> f16 weight conversion (grid-stride, float4 -> v4h) and
// zero-init of the two loss accumulators in workspace.
// ---------------------------------------------------------------------------
__global__ void convert_weights(const float* __restrict__ sw,
                                const float* __restrict__ tw,
                                _Float16* __restrict__ swh,
                                _Float16* __restrict__ twh,
                                float* __restrict__ acc) {
  size_t i      = (size_t)blockIdx.x * blockDim.x + threadIdx.x;
  size_t stride = (size_t)gridDim.x * blockDim.x;
  const size_t NS4 = (size_t)VOC * HS / 4;
  const size_t NT4 = (size_t)VOC * HT / 4;
  for (size_t k = i; k < NS4; k += stride) {
    float4 f = ((const float4*)sw)[k];
    v4h h = { (_Float16)f.x, (_Float16)f.y, (_Float16)f.z, (_Float16)f.w };
    ((v4h*)swh)[k] = h;
  }
  for (size_t k = i; k < NT4; k += stride) {
    float4 f = ((const float4*)tw)[k];
    v4h h = { (_Float16)f.x, (_Float16)f.y, (_Float16)f.z, (_Float16)f.w };
    ((v4h*)twh)[k] = h;
  }
  if (blockIdx.x == 0 && threadIdx.x == 0) { acc[0] = 0.0f; acc[1] = 0.0f; }
}

// ---------------------------------------------------------------------------
// Kernel 2: fused dual-GEMM + online softmax-CE + cosine stats.
// One workgroup per 16-row tile; full-K activations staged in LDS (~197 KB,
// CDNA5 320KB WGP LDS); 8 waves each own PAIRS of vocab tiles (16x32 output
// per pass -> two independent WMMA accumulator chains, A reused from LDS).
// ---------------------------------------------------------------------------
__global__ __launch_bounds__(THREADS)
void fused_distill(const float* __restrict__ student,     // [BT][HS] f32
                   const float* __restrict__ teacher,     // [BT][HT] f32
                   const int*   __restrict__ target,      // [BT]
                   const _Float16* __restrict__ swh,      // [VOC][HS] f16
                   const _Float16* __restrict__ twh,      // [VOC][HT] f16
                   float* __restrict__ acc)               // [0]=hard [1]=soft
{
  extern __shared__ uint4 smem_u4[];               // 16B-aligned dynamic LDS
  _Float16* sA = (_Float16*)smem_u4;               // ROWS x SROW
  _Float16* tA = sA + ROWS * SROW;                 // ROWS x TROW
  float* redbuf = (float*)(tA + ROWS * TROW);      // WAVES*ROWS*6 floats

  const int tid  = threadIdx.x;
  const int lane = tid & 31;
  const int wave = tid >> 5;
  const int r0   = blockIdx.x * ROWS;

  // ---- Stage the 16-row activation tiles into LDS as f16 (vectorized) ----
  for (int idx = tid; idx < ROWS * HS / 4; idx += THREADS) {
    int r = idx >> 9;                  // HS/4 = 512
    int c = (idx & 511) << 2;
    float4 f = *(const float4*)(student + (size_t)(r0 + r) * HS + c);
    v4h h = { (_Float16)f.x, (_Float16)f.y, (_Float16)f.z, (_Float16)f.w };
    *(v4h*)(sA + r * SROW + c) = h;
  }
  for (int idx = tid; idx < ROWS * HT / 4; idx += THREADS) {
    int r = idx >> 10;                 // HT/4 = 1024
    int c = (idx & 1023) << 2;
    float4 f = *(const float4*)(teacher + (size_t)(r0 + r) * HT + c);
    v4h h = { (_Float16)f.x, (_Float16)f.y, (_Float16)f.z, (_Float16)f.w };
    *(v4h*)(tA + r * TROW + c) = h;
  }
  __syncthreads();

  // WMMA lane roles (16x16x32 f16):
  //  A operand : lane holds row m = lane&15 ; half h=lane>>4 selects K-subset
  //              {8h..8h+7} U {16+8h..16+8h+7}  -> two contiguous b128 chunks
  //  B operand : lane holds vocab column n = lane&15 (a row of W), same K split
  //  D matrix  : lane holds column n = lane&15, VGPR j -> row m = j + 8h
  const int h = lane >> 4;
  const int n = lane & 15;

  int tg[8];
#pragma unroll
  for (int j = 0; j < 8; ++j) tg[j] = target[r0 + 8 * h + j];

  float mx[8], sm[8], dt[8], ns[8], nt[8], tl[8];
#pragma unroll
  for (int j = 0; j < 8; ++j) {
    mx[j] = -3.0e38f; sm[j] = 0.f; dt[j] = 0.f; ns[j] = 0.f; nt[j] = 0.f; tl[j] = 0.f;
  }

  const int NTILES = VOC / 16;       // 2000
  const int TSTRIDE = WAVES * 2;     // 16 tiles consumed per block pass
  // Each wave owns tile pair (vt, vt+1): 2000 / 16 = 125 passes exactly.
  for (int vt = wave * 2; vt < NTILES; vt += TSTRIDE) {
    const int vb0 = vt * 16;         // columns of first tile
    const int vb1 = vb0 + 16;        // columns of second tile
    const _Float16* wrs0 = swh + (size_t)(vb0 + n) * HS;
    const _Float16* wrs1 = swh + (size_t)(vb1 + n) * HS;
    const _Float16* wrt0 = twh + (size_t)(vb0 + n) * HT;
    const _Float16* wrt1 = twh + (size_t)(vb1 + n) * HT;

    if (vt + TSTRIDE < NTILES) {     // speculative prefetch of next pass rows
      __builtin_prefetch(swh + (size_t)(vb0 + 16 * TSTRIDE + n) * HS, 0, 1);
      __builtin_prefetch(twh + (size_t)(vb0 + 16 * TSTRIDE + n) * HT, 0, 1);
    }

    // ---- student 16x32 logits tile, K = 2048 : two independent chains ----
    v8f Ds0 = {}, Ds1 = {};
    for (int ks = 0; ks < HS / 32; ++ks) {
      const int kb = ks * 32;
      union { uint4 q[2]; v16h v; } a, b0, b1;
      a.q[0]  = *(const uint4*)(sA + (n * SROW) + kb + 8 * h);
      a.q[1]  = *(const uint4*)(sA + (n * SROW) + kb + 16 + 8 * h);
      b0.q[0] = *(const uint4*)(wrs0 + kb + 8 * h);
      b0.q[1] = *(const uint4*)(wrs0 + kb + 16 + 8 * h);
      b1.q[0] = *(const uint4*)(wrs1 + kb + 8 * h);
      b1.q[1] = *(const uint4*)(wrs1 + kb + 16 + 8 * h);
      Ds0 = __builtin_amdgcn_wmma_f32_16x16x32_f16(false, a.v, false, b0.v,
                                                   (short)0, Ds0, false, false);
      Ds1 = __builtin_amdgcn_wmma_f32_16x16x32_f16(false, a.v, false, b1.v,
                                                   (short)0, Ds1, false, false);
    }
    // ---- teacher 16x32 logits tile, K = 4096 : two independent chains ----
    v8f Dt0 = {}, Dt1 = {};
    for (int ks = 0; ks < HT / 32; ++ks) {
      const int kb = ks * 32;
      union { uint4 q[2]; v16h v; } a, b0, b1;
      a.q[0]  = *(const uint4*)(tA + (n * TROW) + kb + 8 * h);
      a.q[1]  = *(const uint4*)(tA + (n * TROW) + kb + 16 + 8 * h);
      b0.q[0] = *(const uint4*)(wrt0 + kb + 8 * h);
      b0.q[1] = *(const uint4*)(wrt0 + kb + 16 + 8 * h);
      b1.q[0] = *(const uint4*)(wrt1 + kb + 8 * h);
      b1.q[1] = *(const uint4*)(wrt1 + kb + 16 + 8 * h);
      Dt0 = __builtin_amdgcn_wmma_f32_16x16x32_f16(false, a.v, false, b0.v,
                                                   (short)0, Dt0, false, false);
      Dt1 = __builtin_amdgcn_wmma_f32_16x16x32_f16(false, a.v, false, b1.v,
                                                   (short)0, Dt1, false, false);
    }

    // ---- online per-row statistics (each lane: 8 rows, 2 columns) ----
#pragma unroll
    for (int j = 0; j < 8; ++j) {
      float s0 = Ds0[j], t0 = Dt0[j];
      float s1 = Ds1[j], t1 = Dt1[j];
      // online logsumexp over both columns
      float mo = mx[j];
      float mn = fmaxf(mo, fmaxf(s0, s1));
      sm[j] = sm[j] * __expf(mo - mn) + __expf(s0 - mn) + __expf(s1 - mn);
      mx[j] = mn;
      dt[j] = __builtin_fmaf(s0, t0, __builtin_fmaf(s1, t1, dt[j]));
      ns[j] = __builtin_fmaf(s0, s0, __builtin_fmaf(s1, s1, ns[j]));
      nt[j] = __builtin_fmaf(t0, t0, __builtin_fmaf(t1, t1, nt[j]));
      if (vb0 + n == tg[j]) tl[j] += s0;   // target logit (at most one hit)
      if (vb1 + n == tg[j]) tl[j] += s1;
    }
  }

  // ---- reduce across the 16 lanes of each half (xor of low 4 lane bits) ----
#pragma unroll
  for (int off = 1; off < 16; off <<= 1) {
#pragma unroll
    for (int j = 0; j < 8; ++j) {
      float mo = __shfl_xor(mx[j], off);
      float so = __shfl_xor(sm[j], off);
      float mn = fmaxf(mx[j], mo);
      sm[j] = sm[j] * __expf(mx[j] - mn) + so * __expf(mo - mn);
      mx[j] = mn;
      dt[j] += __shfl_xor(dt[j], off);
      ns[j] += __shfl_xor(ns[j], off);
      nt[j] += __shfl_xor(nt[j], off);
      tl[j] += __shfl_xor(tl[j], off);
    }
  }

  if (n == 0) {  // lane 0 (rows 0..7) and lane 16 (rows 8..15)
#pragma unroll
    for (int j = 0; j < 8; ++j) {
      float* p = redbuf + ((wave * ROWS) + (8 * h + j)) * 6;
      p[0] = mx[j]; p[1] = sm[j]; p[2] = dt[j];
      p[3] = ns[j]; p[4] = nt[j]; p[5] = tl[j];
    }
  }
  __syncthreads();

  // ---- merge 8 waves per row, emit per-row loss contributions ----
  if (tid < ROWS) {
    float M = -3.0e38f, S = 0.f, D = 0.f, NS = 0.f, NT = 0.f, T = 0.f;
    for (int w = 0; w < WAVES; ++w) {
      const float* p = redbuf + (w * ROWS + tid) * 6;
      float mo = p[0], so = p[1];
      float mn = fmaxf(M, mo);
      S = S * __expf(M - mn) + so * __expf(mo - mn);
      M = mn;
      D += p[2]; NS += p[3]; NT += p[4]; T += p[5];
    }
    float lse  = M + __logf(S);
    int   tgi  = target[r0 + tid];
    float hard = (tgi != IGNORE_INDEX) ? (lse - T) : 0.0f;
    float cs   = D * rsqrtf(NS + 1e-12f) * rsqrtf(NT + 1e-12f);
    atomicAdd(&acc[0], hard);
    atomicAdd(&acc[1], 1.0f - cs);
  }
}

// ---------------------------------------------------------------------------
// Kernel 3: count valid targets and combine the weighted losses.
// ---------------------------------------------------------------------------
__global__ void finalize_loss(const int* __restrict__ target,
                              const float* __restrict__ acc,
                              float* __restrict__ out) {
  __shared__ int cnt[256];
  int c = 0;
  for (int i = threadIdx.x; i < BT; i += 256)
    c += (target[i] != IGNORE_INDEX) ? 1 : 0;
  cnt[threadIdx.x] = c;
  __syncthreads();
  for (int s = 128; s > 0; s >>= 1) {
    if (threadIdx.x < s) cnt[threadIdx.x] += cnt[threadIdx.x + s];
    __syncthreads();
  }
  if (threadIdx.x == 0) {
    float nv = (float)cnt[0];
    out[0] = (0.5f * acc[0] + 0.5f * acc[1]) / nv;
  }
}

// ---------------------------------------------------------------------------
extern "C" void kernel_launch(void* const* d_in, const int* in_sizes, int n_in,
                              void* d_out, int out_size, void* d_ws, size_t ws_size,
                              hipStream_t stream) {
  const float* student = (const float*)d_in[0];   // [BT, HS]
  const float* teacher = (const float*)d_in[1];   // [BT, HT]
  const int*   target  = (const int*)d_in[2];     // [BT]
  const float* sw      = (const float*)d_in[3];   // [VOC, HS]
  const float* tw      = (const float*)d_in[4];   // [VOC, HT]

  _Float16* swh = (_Float16*)d_ws;                       // 131,072,000 B
  _Float16* twh = swh + (size_t)VOC * HS;                // 262,144,000 B
  float*    acc = (float*)(twh + (size_t)VOC * HT);      // 2 floats

  convert_weights<<<2048, 256, 0, stream>>>(sw, tw, swh, twh, acc);

  size_t lds_bytes = (size_t)(ROWS * SROW + ROWS * TROW) * sizeof(_Float16)
                   + (size_t)WAVES * ROWS * 6 * sizeof(float);   // ~200 KB
  fused_distill<<<BT / ROWS, THREADS, lds_bytes, stream>>>(
      student, teacher, target, swh, twh, acc);

  finalize_loss<<<1, 256, 0, stream>>>(target, acc, (float*)d_out);
}
